// MultiHeadAttentionQuantum_65481071410122
// MI455X (gfx1250) — matrix operand order repaired and checked by reading, not compiled, not run on hardware
//
#include <hip/hip_runtime.h>
#include <hip/hip_bf16.h>

typedef __attribute__((ext_vector_type(16))) _Float16 v16h;
typedef __attribute__((ext_vector_type(8)))  _Float16 v8h;
typedef __attribute__((ext_vector_type(8)))  float    v8f;
typedef __attribute__((ext_vector_type(4)))  unsigned int v4u;
typedef __attribute__((ext_vector_type(8)))  int      v8i;
typedef __attribute__((ext_vector_type(4)))  int      v4i;

constexpr int B_  = 8;
constexpr int S_  = 1024;
constexpr int E_  = 1024;
constexpr int H_  = 16;
constexpr int DK_ = 64;
constexpr int MROWS_ = B_ * S_;   // 8192

#if defined(__has_builtin)
#if __has_builtin(__builtin_amdgcn_tensor_load_to_lds)
#define USE_TDM 1
#endif
#endif

// ---------------------------------------------------------------------------
// WMMA fragment loaders for 16x16x32 f16, contiguous k order (k = 16*half + e) in both A and B
// ---------------------------------------------------------------------------

// A-matrix 16x32 (MxK), row-major source, leading dim `ld` (halves).
// lane m = lane&15, half = lane>>4:
//   h[0..15] = row[half*16 + 0..15]
__device__ inline v16h load_a_frag(const _Float16* base, int ld, int lane) {
  const int m = lane & 15, hf = lane >> 4;
  const _Float16* row = base + (size_t)m * ld;
  v8h lo = *(const v8h*)(row + hf * 16);
  v8h hi = *(const v8h*)(row + hf * 16 + 8);
  v16h a;
#pragma unroll
  for (int i = 0; i < 8; i++) { a[i] = lo[i]; a[8 + i] = hi[i]; }
  return a;
}

// B-matrix 32x16 (KxN) where B[k][n] = src[n*ld + k] (per-lane contiguous in k).
// lane n = lane&15, half = lane>>4: h[e] = src[n*ld + half*16 + e]
__device__ inline v16h load_b_frag_rm(const _Float16* base, int ld, int lane) {
  const int nn = lane & 15, hf = lane >> 4;
  return *(const v16h*)(base + (size_t)nn * ld + hf * 16);
}

// ---------------------------------------------------------------------------
// Kernel A: proj = cos(x + theta), written f16 in [B,H,S,DK] layout.
// ---------------------------------------------------------------------------
__global__ void __launch_bounds__(256)
qproj_cos_kernel(const float* __restrict__ x, const float* __restrict__ theta,
                 _Float16* __restrict__ p) {
  const int idx = blockIdx.x * 256 + threadIdx.x;   // over B*S*E exactly
  const int e  = idx & (E_ - 1);
  const int bs = idx >> 10;             // b*S + s
  const int d  = e & (DK_ - 1);
  const int h  = e >> 6;
  const int b  = bs >> 10;
  const int s  = bs & (S_ - 1);
  const float pr = __cosf(x[idx] + theta[d]);
  p[(((size_t)(b * H_ + h)) * S_ + s) * DK_ + d] = (_Float16)pr;
}

__global__ void __launch_bounds__(256)
wconv_kernel(const float* __restrict__ w, _Float16* __restrict__ wh) {
  const int idx = blockIdx.x * 256 + threadIdx.x;   // over E*E exactly
  wh[idx] = (_Float16)w[idx];
}

// ---------------------------------------------------------------------------
// Kernel B: flash attention, q=k=v=proj = cos(..) in [-1,1].
// |score| = |q.k|/sqrt(64) <= 64/8 = 8, so exp(score) <= e^8 ~ 2981 and the
// row sum <= 3.1e6: softmax is computed max-free (no cross-lane reductions).
// Row sums are produced by one extra WMMA against an all-ones B matrix so
// the l statistics land in the same C/D layout as the output accumulators.
// 4 waves/block; wave owns 16 query rows; 32-key blocks staged in LDS
// (row-major via TDM when available, plus a pre-transposed copy for P.V).
// ---------------------------------------------------------------------------
__global__ void __launch_bounds__(128)
qattn_flash_kernel(const _Float16* __restrict__ p, _Float16* __restrict__ out) {
  __shared__ __align__(32) _Float16 lds_kv [32 * DK_];   // 32 keys x 64 d (row-major)
  __shared__ __align__(32) _Float16 lds_kvT[DK_ * 32];   // 64 d x 32 keys (transposed)
  __shared__ __align__(32) _Float16 lds_p[4][16 * 32];   // per-wave P tile

  const int lane = threadIdx.x & 31;
  const int wave = threadIdx.x >> 5;
  const int nn   = lane & 15;
  const int hf   = lane >> 4;

  const int bid  = blockIdx.x;          // grid = B*H*(S/64) = 2048
  const int qblk = bid & 15;            // S/64 = 16
  const int h    = (bid >> 4) & (H_ - 1);
  const int b    = bid >> 8;
  const _Float16* pbh = p + ((size_t)(b * H_ + h)) * S_ * DK_;
  const int q0 = qblk * 64 + wave * 16;

  // Q fragments: d-slices [0,32) and [32,64)
  const v16h qa0 = load_a_frag(pbh + (size_t)q0 * DK_,      DK_, lane);
  const v16h qa1 = load_a_frag(pbh + (size_t)q0 * DK_ + 32, DK_, lane);

  v16h ones;
#pragma unroll
  for (int i = 0; i < 16; i++) ones[i] = (_Float16)1.0f;

  v8f acc0 = {}, acc1 = {}, acc2 = {}, acc3 = {};
  v8f lacc = {};                          // row sums, C/D layout

  const int crow  = threadIdx.x >> 2;     // staging: 128 thr -> 32 rows x 4 chunks
  const int cpart = threadIdx.x & 3;

  for (int kb = 0; kb < S_; kb += 32) {
    __syncthreads();
    {
      const _Float16* src = pbh + (size_t)(kb + crow) * DK_ + cpart * 16;
      v8h d0 = *(const v8h*)src;
      v8h d1 = *(const v8h*)(src + 8);
      // transposed copy: element j has d = cpart*16+j, key = crow
#pragma unroll
      for (int j = 0; j < 8; j++)
        lds_kvT[(size_t)(cpart * 16 + j) * 32 + crow] = d0[j];
#pragma unroll
      for (int j = 0; j < 8; j++)
        lds_kvT[(size_t)(cpart * 16 + 8 + j) * 32 + crow] = d1[j];
#if defined(USE_TDM)
      if (wave == 0) {
        // Tensor DMA descriptor (CDNA5 ISA):
        // 2D tile: 64 x 32 elements, data_size=2B, row stride 64 elements.
        const unsigned long long ga =
            (unsigned long long)(pbh + (size_t)kb * DK_);
        v4u g0;
        g0[0] = 1u;                                   // count=1 valid descriptor
        g0[1] = (unsigned)(unsigned long long)(lds_kv); // lds_addr (bytes)
        g0[2] = (unsigned)ga;                         // global_addr[31:0]
        g0[3] = ((unsigned)(ga >> 32) & 0x1FFFFFFu)   // global_addr[56:32]
              | (2u << 30);                           // type=2 ("image")
        v8i g1;
        g1[0] = 0x00010000;          // workgroup_mask=0, data_size=1 (2B)
        g1[1] = (int)(64u << 16);    // tensor_dim0 = 64 (low 16 in [31:16])
        g1[2] = (int)(32u << 16);    // tensor_dim0 hi=0 | tensor_dim1 = 32
        g1[3] = (int)(64u << 16);    // tensor_dim1 hi=0 | tile_dim0 = 64
        g1[4] = 32;                  // tile_dim1 = 32, tile_dim2 = 0
        g1[5] = 64;                  // tensor_dim0_stride = 64 (low 32)
        g1[6] = 0;                   // stride hi | tensor_dim1_stride lo
        g1[7] = 0;
        v4i zz = {0, 0, 0, 0};
        v8i z8 = {0, 0, 0, 0, 0, 0, 0, 0};
        // 6-arg variant (clang-23 / therock headers): groups 2,3 + extra group
        __builtin_amdgcn_tensor_load_to_lds(g0, g1, zz, zz, z8, 0);
        __builtin_amdgcn_s_wait_tensorcnt(0);
      }
#else
      _Float16* dst = lds_kv + (size_t)crow * DK_ + cpart * 16;
      *(v8h*)dst       = d0;
      *(v8h*)(dst + 8) = d1;
#endif
      if ((kb + 32 < S_) && (cpart == 0))
        __builtin_prefetch(pbh + (size_t)(kb + 32 + crow) * DK_, 0, 0);
    }
    __syncthreads();

    // scores: two 16x16 tiles (keys kb..+15, kb+16..+31), K=32 over DK twice
    v8f s0 = {}, s1 = {};
    {
      v16h bk = load_b_frag_rm(lds_kv, DK_, lane);
      s0 = __builtin_amdgcn_wmma_f32_16x16x32_f16(false, qa0, false, bk, (short)0, s0, false, false);
      bk = load_b_frag_rm(lds_kv + 32, DK_, lane);
      s0 = __builtin_amdgcn_wmma_f32_16x16x32_f16(false, qa1, false, bk, (short)0, s0, false, false);
      bk = load_b_frag_rm(lds_kv + 16 * DK_, DK_, lane);
      s1 = __builtin_amdgcn_wmma_f32_16x16x32_f16(false, qa0, false, bk, (short)0, s1, false, false);
      bk = load_b_frag_rm(lds_kv + 16 * DK_ + 32, DK_, lane);
      s1 = __builtin_amdgcn_wmma_f32_16x16x32_f16(false, qa1, false, bk, (short)0, s1, false, false);
    }

    // max-free softmax numerator (|score| <= 8): P = exp(score/8), D->A via LDS
    const float sc = 0.125f;  // 1/sqrt(DK)
#pragma unroll
    for (int r = 0; r < 8; r++) {
      const float p0 = __expf(s0[r] * sc);
      const float p1 = __expf(s1[r] * sc);
      lds_p[wave][(r + 8 * hf) * 32 + nn]      = (_Float16)p0;
      lds_p[wave][(r + 8 * hf) * 32 + 16 + nn] = (_Float16)p1;
    }
    asm volatile("s_wait_dscnt 0" ::: "memory");  // in-wave LDS RAW (D->A transpose)

    const v16h pa = load_a_frag(&lds_p[wave][0], 32, lane);
    // row sums via tensor core: lacc += P x ones  (same C/D layout as acc)
    lacc = __builtin_amdgcn_wmma_f32_16x16x32_f16(false, pa, false, ones, (short)0, lacc, false, false);
    // P.V from the pre-transposed tile: contiguous b128 LDS reads per lane
    v16h vb = load_b_frag_rm(lds_kvT + 0 * 16 * 32, 32, lane);
    acc0 = __builtin_amdgcn_wmma_f32_16x16x32_f16(false, pa, false, vb, (short)0, acc0, false, false);
    vb = load_b_frag_rm(lds_kvT + 1 * 16 * 32, 32, lane);
    acc1 = __builtin_amdgcn_wmma_f32_16x16x32_f16(false, pa, false, vb, (short)0, acc1, false, false);
    vb = load_b_frag_rm(lds_kvT + 2 * 16 * 32, 32, lane);
    acc2 = __builtin_amdgcn_wmma_f32_16x16x32_f16(false, pa, false, vb, (short)0, acc2, false, false);
    vb = load_b_frag_rm(lds_kvT + 3 * 16 * 32, 32, lane);
    acc3 = __builtin_amdgcn_wmma_f32_16x16x32_f16(false, pa, false, vb, (short)0, acc3, false, false);
  }

  // normalize + store to [B,S,E] f16 (GEMM A operand layout)
  const size_t obase = ((size_t)b * S_ + q0) * E_ + h * DK_;
#pragma unroll
  for (int r = 0; r < 8; r++) {
    const float inv = 1.0f / lacc[r];
    const size_t o = obase + (size_t)(r + 8 * hf) * E_;
    out[o + 0  + nn] = (_Float16)(acc0[r] * inv);
    out[o + 16 + nn] = (_Float16)(acc1[r] * inv);
    out[o + 32 + nn] = (_Float16)(acc2[r] * inv);
    out[o + 48 + nn] = (_Float16)(acc3[r] * inv);
  }
}

// ---------------------------------------------------------------------------
// Kernel C: out[M,E] = A[M,E] @ W^T  (B[k][n] = W[n][k], rows contiguous in k)
// Wave computes 32x64 tile; block = 8 waves (2M x 4N) -> 64M x 256N tile.
// All operands are L2-resident (A 16MB + W 2MB << 192MB L2).
// ---------------------------------------------------------------------------
__global__ void __launch_bounds__(256)
combine_gemm_kernel(const _Float16* __restrict__ A, const _Float16* __restrict__ Wh,
                    float* __restrict__ out) {
  const int lane = threadIdx.x & 31;
  const int wave = threadIdx.x >> 5;
  const int nn = lane & 15, hf = lane >> 4;
  const int wm = wave & 1, wn = wave >> 1;
  const int m0 = blockIdx.x * 64 + wm * 32;
  const int n0 = blockIdx.y * 256 + wn * 64;

  v8f zacc = {};
  v8f acc[2][4];
#pragma unroll
  for (int i = 0; i < 2; i++)
#pragma unroll
    for (int t = 0; t < 4; t++) acc[i][t] = zacc;

  for (int ks = 0; ks < E_; ks += 32) {
    const v16h a0 = load_a_frag(A + (size_t)m0 * E_ + ks, E_, lane);
    const v16h a1 = load_a_frag(A + (size_t)(m0 + 16) * E_ + ks, E_, lane);
#pragma unroll
    for (int t = 0; t < 4; t++) {
      const v16h bf = load_b_frag_rm(Wh + (size_t)(n0 + t * 16) * E_ + ks, E_, lane);
      acc[0][t] = __builtin_amdgcn_wmma_f32_16x16x32_f16(false, a0, false, bf, (short)0, acc[0][t], false, false);
      acc[1][t] = __builtin_amdgcn_wmma_f32_16x16x32_f16(false, a1, false, bf, (short)0, acc[1][t], false, false);
    }
  }

#pragma unroll
  for (int i = 0; i < 2; i++)
#pragma unroll
    for (int t = 0; t < 4; t++)
#pragma unroll
      for (int r = 0; r < 8; r++) {
        const int row = m0 + i * 16 + r + 8 * hf;
        const int col = n0 + t * 16 + nn;
        out[(size_t)row * E_ + col] = acc[i][t][r];
      }
}

// ---------------------------------------------------------------------------
extern "C" void kernel_launch(void* const* d_in, const int* in_sizes, int n_in,
                              void* d_out, int out_size, void* d_ws, size_t ws_size,
                              hipStream_t stream) {
  (void)in_sizes; (void)n_in; (void)out_size; (void)ws_size;
  const float* x     = (const float*)d_in[0];   // [B,S,E]
  const float* theta = (const float*)d_in[1];   // [DK]
  const float* W     = (const float*)d_in[2];   // [E,E]
  float* out = (float*)d_out;                   // [B,S,E]

  const size_t pBytes    = (size_t)B_ * S_ * E_ * sizeof(_Float16);   // 16 MB
  const size_t attnBytes = pBytes;                                    // 16 MB
  _Float16* p     = (_Float16*)d_ws;                                  // [B,H,S,DK] f16
  _Float16* attnb = (_Float16*)((char*)d_ws + pBytes);                // [B*S, E] f16
  _Float16* wh    = (_Float16*)((char*)d_ws + pBytes + attnBytes);    // [E, E]   f16

  qproj_cos_kernel<<<(B_ * S_ * E_) / 256, 256, 0, stream>>>(x, theta, p);
  wconv_kernel<<<(E_ * E_) / 256, 256, 0, stream>>>(W, wh);
  qattn_flash_kernel<<<B_ * H_ * (S_ / 64), 128, 0, stream>>>(p, attnb);
  combine_gemm_kernel<<<dim3(MROWS_ / 64, E_ / 256), 256, 0, stream>>>(attnb, wh, out);
}
